// MultiHeadAttention_32796370272591
// MI455X (gfx1250) — matrix operand order repaired and checked by reading, not compiled, not run on hardware
//
#include <hip/hip_runtime.h>

// ---------------- CDNA5 WMMA types ----------------
typedef __bf16 v16bf __attribute__((ext_vector_type(16)));
typedef __bf16 bf16x8 __attribute__((ext_vector_type(8)));
typedef float  v8f    __attribute__((ext_vector_type(8)));
typedef float  f32x4  __attribute__((ext_vector_type(4)));

union FragBF {
    v16bf  v;
    bf16x8 h[2];
    __bf16 e[16];
};

__device__ __forceinline__ v8f wmma_bf16(v16bf a, v16bf b, v8f c) {
    // v_wmma_f32_16x16x32_bf16  D = A*B + C, f32 accumulate
    return __builtin_amdgcn_wmma_f32_16x16x32_bf16(
        /*neg_a=*/false, a, /*neg_b=*/false, b,
        /*c_mod=*/(short)0, c, /*reuse_a=*/false, /*reuse_b=*/false);
}

// A operand (16x32, 16-bit): lane half hh holds K = hh*8..+7 and 16+hh*8..+7
__device__ __forceinline__ v16bf load_a_bf16(const __bf16* row, int kb, int hh) {
    FragBF f;
    f.h[0] = *(const bf16x8*)(row + kb + hh * 8);
    f.h[1] = *(const bf16x8*)(row + kb + 16 + hh * 8);
    return f.v;
}

// A operand built from an fp32 row (global or LDS) with on-the-fly bf16 cvt
__device__ __forceinline__ v16bf load_a_f32(const float* row, int kb, int hh) {
    f32x4 x0 = *(const f32x4*)(row + kb + hh * 8);
    f32x4 x1 = *(const f32x4*)(row + kb + hh * 8 + 4);
    f32x4 x2 = *(const f32x4*)(row + kb + 16 + hh * 8);
    f32x4 x3 = *(const f32x4*)(row + kb + 16 + hh * 8 + 4);
    FragBF f;
#pragma unroll
    for (int i = 0; i < 4; ++i) {
        f.e[i]      = (__bf16)x0[i];
        f.e[4 + i]  = (__bf16)x1[i];
        f.e[8 + i]  = (__bf16)x2[i];
        f.e[12 + i] = (__bf16)x3[i];
    }
    return f.v;
}

// B operand (32x16, 16-bit): lane holds column n = lane&15, K = hh*8..+7 and 16+hh*8..+7
// rowT must point at the (transposed) column, i.e. B^T row n, K-contiguous.
__device__ __forceinline__ v16bf load_b_bf16(const __bf16* rowT, int kb, int hh) {
    FragBF f;
    f.h[0] = *(const bf16x8*)(rowT + kb + hh * 8);
    f.h[1] = *(const bf16x8*)(rowT + kb + 16 + hh * 8);
    return f.v;
}

// order-preserving float<->uint maps for exact radix top-k select
__device__ __forceinline__ unsigned f2ord(float f) {
    unsigned b = __float_as_uint(f);
    return (b & 0x80000000u) ? ~b : (b | 0x80000000u);
}
__device__ __forceinline__ float ord2f(unsigned u) {
    unsigned b = (u & 0x80000000u) ? (u & 0x7fffffffu) : ~u;
    return __uint_as_float(b);
}

// ---------------- problem constants ----------------
#define CB    2
#define CL    2048
#define CE    1024
#define CH    16
#define CDK   64
#define CDV   64
#define CTOPK 64

// ---------------- kernels ----------------

// fp32 [R][C] -> bf16 [C][R]
__global__ void transpose_to_bf16(const float* __restrict__ src,
                                  __bf16* __restrict__ dst, int R, int C) {
    int idx = blockIdx.x * blockDim.x + threadIdx.x;
    if (idx >= R * C) return;
    int r = idx / C, c = idx % C;
    dst[(size_t)c * R + r] = (__bf16)src[idx];
}

// C[M,N] = alpha * A[M,K] @ B[K,N]   (B given transposed: BT[N,K] bf16)
// mode 0: dst bf16 head-split [B,H,L,DK]   (n = h*64+d, m = b*L+l)
// mode 1: dst bf16 head-split transposed [B,H,DV,L]
// mode 2: dst fp32 row-major [M,N]
__global__ void __launch_bounds__(128)
gemm_wmma(const float* __restrict__ A, const __bf16* __restrict__ BT,
          int M, int N, int K, float alpha, int mode, void* __restrict__ dstv) {
    const int lane = threadIdx.x & 31;
    const int wv   = threadIdx.x >> 5;
    const int hh   = lane >> 4;
    const int r16  = lane & 15;

    const int n0 = blockIdx.x * 64 + wv * 16;
    const int m0 = blockIdx.y * 16;
    if (n0 >= N || m0 >= M) return;

    const float*  rowA = A  + (size_t)(m0 + r16) * K;
    const __bf16* rowB = BT + (size_t)(n0 + r16) * K;

    v8f acc = {0.f, 0.f, 0.f, 0.f, 0.f, 0.f, 0.f, 0.f};
    for (int kb = 0; kb < K; kb += 32) {
        __builtin_prefetch(rowA + kb + 64, 0, 1);
        __builtin_prefetch(rowB + kb + 64, 0, 1);
        v16bf a = load_a_f32(rowA, kb, hh);
        v16bf b = load_b_bf16(rowB, kb, hh);
        acc = wmma_bf16(a, b, acc);
    }

#pragma unroll
    for (int rr = 0; rr < 8; ++rr) {
        int m = m0 + hh * 8 + rr;
        int n = n0 + r16;
        float val = acc[rr] * alpha;
        if (mode == 2) {
            ((float*)dstv)[(size_t)m * N + n] = val;
        } else {
            int bb = m >> 11, ll = m & (CL - 1);
            int h  = n >> 6,  d  = n & 63;
            size_t o = (mode == 0)
                ? ((size_t)((bb * CH + h) * CL + ll) * CDK + d)
                : ((size_t)((bb * CH + h) * CDV + d) * CL + ll);
            ((__bf16*)dstv)[o] = (__bf16)val;
        }
    }
}

// Fused attention: logits (WMMA, into LDS) -> exact top-64 mask + softmax
// (radix select over LDS, single fp32 global write) -> attn@V (WMMA from LDS)
// grid: (L/16, H, B), block: 256 (8 waves)
__global__ void __launch_bounds__(256)
attn_topk_kernel(const __bf16* __restrict__ qp, const __bf16* __restrict__ kp,
                 const __bf16* __restrict__ vpT,
                 float* __restrict__ attn_out, float* __restrict__ mixed_ws) {
    __shared__ float logit_lds[16][CL];     // 128 KB: logits, then softmax probs
    __shared__ float mixed_lds[16 * CDV];   // 4 KB:   fp32 partial output

    const int tid  = threadIdx.x;
    const int lane = tid & 31;
    const int wv   = tid >> 5;
    const int hh   = lane >> 4;
    const int r16  = lane & 15;

    const int q0 = blockIdx.x * 16;
    const int h  = blockIdx.y;
    const int b  = blockIdx.z;
    const int bh = b * CH + h;

    for (int i = tid; i < 16 * CDV; i += 256) mixed_lds[i] = 0.f;

    const __bf16* qrow  = qp + ((size_t)bh * CL + q0 + r16) * CDK;
    const __bf16* kbase = kp + (size_t)bh * CL * CDK;
    float*        arow  = attn_out + ((size_t)bh * CL + q0) * CL;

    // ---- phase 1: logits[16 x 2048] = (scaled q) @ k^T, straight into LDS
    v16bf a0 = load_a_bf16(qrow, 0, hh);
    v16bf a1 = load_a_bf16(qrow, 32, hh);
    for (int nt = wv; nt < CL / 16; nt += 8) {
        int n0 = nt * 16;
        const __bf16* krow = kbase + (size_t)(n0 + r16) * CDK;
        v16bf b0 = load_b_bf16(krow, 0, hh);
        v16bf b1 = load_b_bf16(krow, 32, hh);
        v8f c = {0.f, 0.f, 0.f, 0.f, 0.f, 0.f, 0.f, 0.f};
        c = wmma_bf16(a0, b0, c);
        c = wmma_bf16(a1, b1, c);
#pragma unroll
        for (int rr = 0; rr < 8; ++rr)
            logit_lds[hh * 8 + rr][n0 + r16] = c[rr];
    }
    __syncthreads();

    // ---- phase 2: per-row exact kth-largest threshold + masked softmax
    // (all row traffic is LDS, vectorized 4-wide; one coalesced global write)
    for (int mrow = wv; mrow < 16; mrow += 8) {
        float* rowp  = &logit_lds[mrow][0];
        float* growp = arow + (size_t)mrow * CL;

        // row max (always a kept element)
        unsigned um = 0;
#pragma unroll 1
        for (int c = 0; c < CL / 128; ++c) {
            f32x4 x = *(const f32x4*)(rowp + c * 128 + lane * 4);
#pragma unroll
            for (int i = 0; i < 4; ++i) {
                unsigned u = f2ord(x[i]);
                um = um > u ? um : u;
            }
        }
        for (int off = 16; off; off >>= 1) {
            unsigned o = (unsigned)__shfl_xor((int)um, off);
            um = um > o ? um : o;
        }
        float mf = ord2f(um);

        // radix select: largest T with count(u >= T) >= TOP_K  (== kth value)
        unsigned pref = 0;
#pragma unroll 1
        for (int bit = 31; bit >= 0; --bit) {
            unsigned cand = pref | (1u << bit);
            int cnt = 0;
#pragma unroll 1
            for (int c = 0; c < CL / 128; ++c) {
                f32x4 x = *(const f32x4*)(rowp + c * 128 + lane * 4);
#pragma unroll
                for (int i = 0; i < 4; ++i)
                    cnt += (f2ord(x[i]) >= cand) ? 1 : 0;
            }
            for (int off = 16; off; off >>= 1) cnt += __shfl_xor(cnt, off);
            if (cnt >= CTOPK) pref = cand;
        }

        // denominator over kept elements
        float s = 0.f;
#pragma unroll 1
        for (int c = 0; c < CL / 128; ++c) {
            f32x4 x = *(const f32x4*)(rowp + c * 128 + lane * 4);
#pragma unroll
            for (int i = 0; i < 4; ++i)
                if (f2ord(x[i]) >= pref) s += expf(x[i] - mf);
        }
        for (int off = 16; off; off >>= 1) s += __shfl_xor(s, off);
        float invs = 1.0f / s;

        // write masked softmax: fp32 -> global (only attn HBM write)
        //                        fp32 -> LDS in place (phase-3 operand)
#pragma unroll 1
        for (int c = 0; c < CL / 128; ++c) {
            int j = c * 128 + lane * 4;
            f32x4 x = *(const f32x4*)(rowp + j);
            f32x4 p;
#pragma unroll
            for (int i = 0; i < 4; ++i)
                p[i] = (f2ord(x[i]) >= pref) ? expf(x[i] - mf) * invs : 0.0f;
            *(f32x4*)(growp + j) = p;
            *(f32x4*)(rowp + j)  = p;
        }
    }
    __syncthreads();

    // ---- phase 3: mixed[16 x 64] = attn[16 x 2048] @ vp[2048 x 64]
    {
        const int ntile = wv & 3;
        const int khalf = wv >> 2;
        const int n0 = ntile * 16;
        const __bf16* vrow = vpT + ((size_t)bh * CDV + n0 + r16) * CL;
        const float*  asrc = &logit_lds[r16][0];
        v8f acc = {0.f, 0.f, 0.f, 0.f, 0.f, 0.f, 0.f, 0.f};
        for (int ks = 0; ks < 32; ++ks) {
            int kb = khalf * 1024 + ks * 32;
            v16bf a = load_a_f32(asrc, kb, hh);    // ds_load b128 x4 + cvt
            v16bf v = load_b_bf16(vrow, kb, hh);   // global b128 x2
            acc = wmma_bf16(a, v, acc);
        }
#pragma unroll
        for (int rr = 0; rr < 8; ++rr)
            atomicAdd(&mixed_lds[(hh * 8 + rr) * CDV + n0 + r16], acc[rr]);
    }
    __syncthreads();

    // ---- phase 4: mixed -> ws fp32 [B*L, H*DV]
    for (int i = tid; i < 16 * CDV; i += 256) {
        int m = i >> 6, d = i & 63;
        mixed_ws[((size_t)b * CL + q0 + m) * (CH * CDV) + h * CDV + d] = mixed_lds[i];
    }
}

// ---------------- host launcher ----------------
extern "C" void kernel_launch(void* const* d_in, const int* in_sizes, int n_in,
                              void* d_out, int out_size, void* d_ws, size_t ws_size,
                              hipStream_t stream) {
    const float* q    = (const float*)d_in[0];
    const float* k    = (const float*)d_in[1];
    const float* v    = (const float*)d_in[2];
    const float* w_qs = (const float*)d_in[3];
    const float* w_ks = (const float*)d_in[4];
    const float* w_vs = (const float*)d_in[5];
    const float* fc   = (const float*)d_in[6];

    float* out  = (float*)d_out;                       // [B, L, DV]
    float* attn = out + (size_t)CB * CL * CDV;         // [B, H, L, L]

    char* ws = (char*)d_ws;
    const size_t MB = 1ull << 20;
    __bf16* qp_bf = (__bf16*)(ws + 0 * MB);            // [B,H,L,DK] bf16   8 MB
    __bf16* kp_bf = (__bf16*)(ws + 8 * MB);            // [B,H,L,DK] bf16   8 MB
    __bf16* vpT   = (__bf16*)(ws + 16 * MB);           // [B,H,DV,L] bf16   8 MB
    __bf16* wqsT  = (__bf16*)(ws + 24 * MB);           // [1024,1024] bf16  2 MB
    __bf16* wksT  = (__bf16*)(ws + 26 * MB);           // [1024,1024] bf16  2 MB
    __bf16* wvsT  = (__bf16*)(ws + 28 * MB);           // [1024,64]   bf16  128 KB
    __bf16* fcT   = (__bf16*)(ws + 28 * MB + 256 * 1024); // [64,1024] bf16 128 KB
    float*  mixed = (float*)(ws + 29 * MB);            // [B*L, H*DV] fp32  16 MB

    const int M  = CB * CL;           // 4096
    const int HN = CH * CDK;          // 1024

    // weights -> bf16, transposed (B^T layout for WMMA B operand)
    transpose_to_bf16<<<(CE * HN + 255) / 256, 256, 0, stream>>>(w_qs, wqsT, CE, HN);
    transpose_to_bf16<<<(CE * HN + 255) / 256, 256, 0, stream>>>(w_ks, wksT, CE, HN);
    transpose_to_bf16<<<(CDV * (CH * CDV) + 255) / 256, 256, 0, stream>>>(w_vs, wvsT, CDV, CH * CDV);
    transpose_to_bf16<<<((CH * CDV) * CDV + 255) / 256, 256, 0, stream>>>(fc, fcT, CH * CDV, CDV);

    // projections (scale/TEMP = (1/sqrt(64))/0.8 folded into qp)
    gemm_wmma<<<dim3(HN / 64, M / 16), 128, 0, stream>>>(q, wqsT, M, HN, CE, 0.15625f, 0, qp_bf);
    gemm_wmma<<<dim3(HN / 64, M / 16), 128, 0, stream>>>(k, wksT, M, HN, CE, 1.0f, 0, kp_bf);
    gemm_wmma<<<dim3((CH * CDV) / 64, M / 16), 128, 0, stream>>>(v, wvsT, M, CH * CDV, CDV, 1.0f, 1, vpT);

    // fused attention with exact top-64 restriction
    attn_topk_kernel<<<dim3(CL / 16, CH, CB), 256, 0, stream>>>(qp_bf, kp_bf, vpT, attn, mixed);

    // final fc projection -> out (fp32)
    gemm_wmma<<<dim3(1, M / 16), 128, 0, stream>>>(mixed, fcT, M, CDV, CH * CDV, 1.0f, 2, out);
}
